// EncoderDecoder_21655225106595
// MI455X (gfx1250) — matrix operand order repaired and checked by reading, not compiled, not run on hardware
//
#include <hip/hip_runtime.h>
#include <hip/hip_bf16.h>

// ---------------------------------------------------------------------------
// Encoder/decoder GRU for MI455X (gfx1250).
//  * bf16 WMMA (v_wmma_f32_16x16x32_bf16), f32 accumulate.
//  * Weights pre-packed into wave32 B-fragment layout (CDNA5 ISA 7.12.2).
//  * Persistent kernels; grid-wide atomic barrier per recurrent step.
//  * 64 WGs x 256 thr: wave = M-tile (batch/16), WG = 16 hidden columns x 3 gates.
// ---------------------------------------------------------------------------

#define DEVINL __device__ __forceinline__

typedef __bf16 bf16_t;
typedef bf16_t v16bf __attribute__((ext_vector_type(16)));
typedef float  v8f   __attribute__((ext_vector_type(8)));

static constexpr int B_    = 128;
static constexpr int TIN_  = 512;
static constexpr int TOUT_ = 64;
static constexpr int INF_  = 64;
static constexpr int OUTF_ = 32;
static constexpr int H_    = 1024;
static constexpr int NWG   = 64;    // H_/16 column tiles -> one per workgroup
static constexpr int NTHR  = 256;   // 8 waves == 8 M-tiles (B_=128)

DEVINL unsigned short f2bf(float f) {
  unsigned u = __float_as_uint(f);
  u += 0x7fffu + ((u >> 16) & 1u);          // round-to-nearest-even
  return (unsigned short)(u >> 16);
}

DEVINL float sigm(float x) { return 1.0f / (1.0f + __expf(-x)); }

union Frag {
  v16bf v;
  uint4 q[2];
};

DEVINL v8f vzero() {
  v8f v;
#pragma unroll
  for (int i = 0; i < 8; ++i) v[i] = 0.f;
  return v;
}

// device-wide split barrier: monotonically increasing counter, one event/phase
DEVINL void grid_barrier(unsigned* ctr, unsigned& ev) {
  __threadfence();
  __syncthreads();
  ++ev;
  if (threadIdx.x == 0) {
    __hip_atomic_fetch_add(ctr, 1u, __ATOMIC_ACQ_REL, __HIP_MEMORY_SCOPE_AGENT);
    const unsigned target = (unsigned)gridDim.x * ev;
    while (__hip_atomic_load(ctr, __ATOMIC_ACQUIRE, __HIP_MEMORY_SCOPE_AGENT) < target) {
      __builtin_amdgcn_s_sleep(2);
    }
  }
  __syncthreads();
}

// A fragment (16x32 bf16, row-major source): lane m = lane&15,
// K = [8*half, 8*half+8) then [16+8*half, 16+8*half+8)  -> two b128 loads
DEVINL v16bf load_afrag(const unsigned short* base, int ldk, int m0, int k0, int lane) {
  const int m = lane & 15;
  const int half = lane >> 4;
  const unsigned short* p =
      base + (size_t)(m0 + m) * (size_t)ldk + (size_t)(k0 + half * 8);
  Frag f;
  f.q[0] = *reinterpret_cast<const uint4*>(p);
  f.q[1] = *reinterpret_cast<const uint4*>(p + 16);
  return f.v;
}

// B fragment from pre-packed weights: 512 bf16 per 32x16 tile, lane-major.
DEVINL v16bf load_bfrag(const unsigned short* pk, int tile, int lane) {
  const unsigned short* p = pk + (size_t)tile * 512 + (size_t)lane * 16;
  Frag f;
  f.q[0] = *reinterpret_cast<const uint4*>(p);
  f.q[1] = *reinterpret_cast<const uint4*>(p + 8);
  return f.v;
}

DEVINL v8f wmma_bf16(v16bf a, v16bf b, v8f c) {
  return __builtin_amdgcn_wmma_f32_16x16x32_bf16(false, a, false, b, (short)0, c,
                                                 false, false);
}

// 3-gate GEMM strip: acc[g] += A(m-tile, K) * W_g(n-tile w, K)^T
DEVINL void gemm3(v8f* acc, const unsigned short* Abase, int ldk, int m0,
                  const unsigned short* pk, int w, int KT, int lane) {
  for (int kt = 0; kt < KT; ++kt) {
    v16bf a = load_afrag(Abase, ldk, m0, kt * 32, lane);
#pragma unroll
    for (int g = 0; g < 3; ++g) {
      v16bf b = load_bfrag(pk, (g * NWG + w) * KT + kt, lane);
      acc[g] = wmma_bf16(a, b, acc[g]);
    }
  }
}

// GRU gate math on the C/D layout: lane holds (m = m0+8*half+i, col j)
DEVINL void gru_update(const v8f* ai, const v8f* ah,
                       float bir, float biz, float bin_,
                       float bhr, float bhz, float bhn,
                       const float* __restrict__ hin_f,
                       float* __restrict__ hout_f,
                       unsigned short* __restrict__ hout_b,
                       int m0, int j, int lane) {
  const int half = lane >> 4;
#pragma unroll
  for (int i = 0; i < 8; ++i) {
    const int m = m0 + half * 8 + i;
    const float r = sigm(ai[0][i] + bir + ah[0][i] + bhr);
    const float z = sigm(ai[1][i] + biz + ah[1][i] + bhz);
    const float n = tanhf(ai[2][i] + bin_ + r * (ah[2][i] + bhn));
    const float ho = hin_f[(size_t)m * H_ + j];
    const float hn = (1.0f - z) * n + z * ho;
    hout_f[(size_t)m * H_ + j] = hn;
    hout_b[(size_t)m * H_ + j] = f2bf(hn);
  }
}

// ---------------------------------------------------------------------------
// preprocessing kernels
// ---------------------------------------------------------------------------

__global__ void k_init(unsigned* ctr) {
  if (threadIdx.x < 2 && blockIdx.x == 0) ctr[threadIdx.x] = 0;
}

__global__ void k_cast_bf16(const float* __restrict__ in,
                            unsigned short* __restrict__ out, int n) {
  int i = blockIdx.x * blockDim.x + threadIdx.x;
  if (i < n) out[i] = f2bf(in[i]);
}

// Pack row-major W (N x K, f32) into WMMA B-fragment tiles (bf16):
// tile (nt,kt) flat = nt*(K/32)+kt; lane l, elem i <- W[nt*16 + (l&15)][kt*32 + (l>>4)*8 + i + (i&8)]
__global__ void k_pack(const float* __restrict__ W,
                       unsigned short* __restrict__ out, int N, int K) {
  const int tid = blockIdx.x * blockDim.x + threadIdx.x;
  const int ktiles = K / 32;
  const int total = (N / 16) * ktiles * 32;
  if (tid >= total) return;
  const int lane = tid & 31;
  const int tile = tid >> 5;
  const int kt = tile % ktiles;
  const int nt = tile / ktiles;
  const int lj = lane & 15;
  const int half = lane >> 4;
  const float* src = W + (size_t)(nt * 16 + lj) * K + kt * 32 + half * 8;
  unsigned short* dst = out + (size_t)tile * 512 + (size_t)lane * 16;
#pragma unroll
  for (int i = 0; i < 16; ++i) dst[i] = f2bf(src[i + (i & 8)]);
}

// ---------------------------------------------------------------------------
// persistent encoder: both GRU layers fused per timestep
// ---------------------------------------------------------------------------

struct EncParams {
  const unsigned short* xbf;   // enc_input bf16 (B,TIN,64)
  const unsigned short* wih0;  // packed, KT=2
  const unsigned short* whh0;  // packed, KT=32
  const unsigned short* wih1;  // packed, KT=32
  const unsigned short* whh1;  // packed, KT=32
  const float* bih0; const float* bhh0;
  const float* bih1; const float* bhh1;
  float* h0f[2]; float* h1f[2];
  unsigned short* h0b[2]; unsigned short* h1b[2];
  unsigned* ctr;
};

__global__ __launch_bounds__(NTHR) void k_encoder(EncParams p) {
  const int w = blockIdx.x;
  const int lane = threadIdx.x & 31;
  const int wave = threadIdx.x >> 5;
  const int m0 = wave * 16;
  const int j = w * 16 + (lane & 15);
  unsigned ev = 0;

  // zero-init h0/h1 ping buffer
  for (int idx = blockIdx.x * blockDim.x + threadIdx.x; idx < B_ * H_;
       idx += gridDim.x * blockDim.x) {
    p.h0f[0][idx] = 0.f; p.h1f[0][idx] = 0.f;
    p.h0b[0][idx] = 0;   p.h1b[0][idx] = 0;
  }
  grid_barrier(p.ctr, ev);

  const float b0ir = p.bih0[j], b0iz = p.bih0[H_ + j], b0in = p.bih0[2 * H_ + j];
  const float b0hr = p.bhh0[j], b0hz = p.bhh0[H_ + j], b0hn = p.bhh0[2 * H_ + j];
  const float b1ir = p.bih1[j], b1iz = p.bih1[H_ + j], b1in = p.bih1[2 * H_ + j];
  const float b1hr = p.bhh1[j], b1hz = p.bhh1[H_ + j], b1hn = p.bhh1[2 * H_ + j];

  for (int t = 0; t < TIN_; ++t) {
    const int ci = t & 1, co = ci ^ 1;
    {  // layer-0 cell: gi (K=64, fused) + gh (K=1024)
      v8f ai[3] = {vzero(), vzero(), vzero()};
      v8f ah[3] = {vzero(), vzero(), vzero()};
      gemm3(ai, p.xbf + (size_t)t * INF_, TIN_ * INF_, m0, p.wih0, w, INF_ / 32, lane);
      gemm3(ah, p.h0b[ci], H_, m0, p.whh0, w, H_ / 32, lane);
      gru_update(ai, ah, b0ir, b0iz, b0in, b0hr, b0hz, b0hn,
                 p.h0f[ci], p.h0f[co], p.h0b[co], m0, j, lane);
    }
    grid_barrier(p.ctr, ev);
    {  // layer-1 cell: input = y0[t] = fresh h0
      v8f ai[3] = {vzero(), vzero(), vzero()};
      v8f ah[3] = {vzero(), vzero(), vzero()};
      gemm3(ai, p.h0b[co], H_, m0, p.wih1, w, H_ / 32, lane);
      gemm3(ah, p.h1b[ci], H_, m0, p.whh1, w, H_ / 32, lane);
      gru_update(ai, ah, b1ir, b1iz, b1in, b1hr, b1hz, b1hn,
                 p.h1f[ci], p.h1f[co], p.h1b[co], m0, j, lane);
    }
    grid_barrier(p.ctr, ev);
  }
  // TIN_ even -> final states land in buffer 0, where the decoder starts.
}

// ---------------------------------------------------------------------------
// persistent decoder: 2 GRU cells + relu(out1) + out2 per step, autoregressive
// ---------------------------------------------------------------------------

struct DecParams {
  const float* enc_in;          // f32, for y_prev0 = enc_input[:, -1, :32]
  const unsigned short* xbf;    // dec_input bf16 (B,TOUT,64)
  const unsigned short* wih0;   // packed, KT=3 (DEC_IN=96)
  const unsigned short* whh0;   // KT=32
  const unsigned short* wih1;   // KT=32
  const unsigned short* whh1;   // KT=32
  const unsigned short* o1pk;   // out1_W packed (64 n-tiles, KT=32)
  const unsigned short* o2pk;   // out2_W packed (2 n-tiles, KT=32)
  const float* bih0; const float* bhh0;
  const float* bih1; const float* bhh1;
  const float* o1b;
  float* h0f[2]; float* h1f[2];
  unsigned short* h0b[2]; unsigned short* h1b[2];
  unsigned short* obf;          // relu output, bf16 (B,H)
  unsigned short* ypb;          // y_prev, bf16 (B,32)
  float* out;                   // (B,TOUT,32) f32
  unsigned* ctr;
};

__global__ __launch_bounds__(NTHR) void k_decoder(DecParams p) {
  const int w = blockIdx.x;
  const int lane = threadIdx.x & 31;
  const int wave = threadIdx.x >> 5;
  const int m0 = wave * 16;
  const int j = w * 16 + (lane & 15);
  unsigned ev = 0;

  {  // y_prev0
    int gid = blockIdx.x * blockDim.x + threadIdx.x;
    if (gid < B_ * OUTF_) {
      int b = gid >> 5, f = gid & 31;
      p.ypb[gid] =
          f2bf(p.enc_in[(size_t)b * TIN_ * INF_ + (size_t)(TIN_ - 1) * INF_ + f]);
    }
  }
  grid_barrier(p.ctr, ev);

  const float d0ir = p.bih0[j], d0iz = p.bih0[H_ + j], d0in = p.bih0[2 * H_ + j];
  const float d0hr = p.bhh0[j], d0hz = p.bhh0[H_ + j], d0hn = p.bhh0[2 * H_ + j];
  const float d1ir = p.bih1[j], d1iz = p.bih1[H_ + j], d1in = p.bih1[2 * H_ + j];
  const float d1hr = p.bhh1[j], d1hz = p.bhh1[H_ + j], d1hn = p.bhh1[2 * H_ + j];
  const float o1bj = p.o1b[j];

  for (int t = 0; t < TOUT_; ++t) {
    const int ci = t & 1, co = ci ^ 1;
    {  // cell 0: x = [y_prev(32) | dec_x_t(64)], K = 96
      v8f ai[3] = {vzero(), vzero(), vzero()};
      v8f ah[3] = {vzero(), vzero(), vzero()};
      {  // k-tile 0: y_prev
        v16bf a = load_afrag(p.ypb, OUTF_, m0, 0, lane);
#pragma unroll
        for (int g = 0; g < 3; ++g)
          ai[g] = wmma_bf16(a, load_bfrag(p.wih0, (g * NWG + w) * 3, lane), ai[g]);
      }
      for (int kt = 1; kt < 3; ++kt) {  // k-tiles 1..2: dec_input cols 0..63
        v16bf a = load_afrag(p.xbf + (size_t)t * INF_ - 32, TOUT_ * INF_, m0,
                             kt * 32, lane);
#pragma unroll
        for (int g = 0; g < 3; ++g)
          ai[g] =
              wmma_bf16(a, load_bfrag(p.wih0, (g * NWG + w) * 3 + kt, lane), ai[g]);
      }
      gemm3(ah, p.h0b[ci], H_, m0, p.whh0, w, H_ / 32, lane);
      gru_update(ai, ah, d0ir, d0iz, d0in, d0hr, d0hz, d0hn,
                 p.h0f[ci], p.h0f[co], p.h0b[co], m0, j, lane);
    }
    grid_barrier(p.ctr, ev);
    {  // cell 1
      v8f ai[3] = {vzero(), vzero(), vzero()};
      v8f ah[3] = {vzero(), vzero(), vzero()};
      gemm3(ai, p.h0b[co], H_, m0, p.wih1, w, H_ / 32, lane);
      gemm3(ah, p.h1b[ci], H_, m0, p.whh1, w, H_ / 32, lane);
      gru_update(ai, ah, d1ir, d1iz, d1in, d1hr, d1hz, d1hn,
                 p.h1f[ci], p.h1f[co], p.h1b[co], m0, j, lane);
    }
    grid_barrier(p.ctr, ev);
    {  // out1: o = relu(h1 @ out1_W^T + b1); WG w owns n-tile w of 64
      v8f acc = vzero();
      for (int kt = 0; kt < H_ / 32; ++kt) {
        v16bf a = load_afrag(p.h1b[co], H_, m0, kt * 32, lane);
        acc = wmma_bf16(a, load_bfrag(p.o1pk, w * 32 + kt, lane), acc);
      }
      const int half = lane >> 4;
#pragma unroll
      for (int i = 0; i < 8; ++i) {
        const int m = m0 + half * 8 + i;
        float v = acc[i] + o1bj;
        v = v > 0.f ? v : 0.f;
        p.obf[(size_t)m * H_ + j] = f2bf(v);
      }
    }
    grid_barrier(p.ctr, ev);
    if (w < 2) {  // out2: y = o @ out2_W^T, N = 32 -> WGs 0,1
      v8f acc = vzero();
      for (int kt = 0; kt < H_ / 32; ++kt) {
        v16bf a = load_afrag(p.obf, H_, m0, kt * 32, lane);
        acc = wmma_bf16(a, load_bfrag(p.o2pk, w * 32 + kt, lane), acc);
      }
      const int half = lane >> 4;
#pragma unroll
      for (int i = 0; i < 8; ++i) {
        const int b = m0 + half * 8 + i;
        const int f = j;  // w<2 => j in [0,32)
        const float y = acc[i];
        p.out[(size_t)b * TOUT_ * OUTF_ + (size_t)t * OUTF_ + f] = y;
        p.ypb[b * OUTF_ + f] = f2bf(y);
      }
    }
    grid_barrier(p.ctr, ev);
  }
}

// ---------------------------------------------------------------------------
// host launcher
// ---------------------------------------------------------------------------

extern "C" void kernel_launch(void* const* d_in, const int* in_sizes, int n_in,
                              void* d_out, int out_size, void* d_ws, size_t ws_size,
                              hipStream_t stream) {
  (void)in_sizes; (void)n_in; (void)out_size; (void)ws_size;

  const float* enc_input = (const float*)d_in[0];
  const float* dec_input = (const float*)d_in[1];
  const float* enc_Wih0  = (const float*)d_in[2];
  const float* enc_Whh0  = (const float*)d_in[3];
  const float* enc_bih0  = (const float*)d_in[4];
  const float* enc_bhh0  = (const float*)d_in[5];
  const float* enc_Wih1  = (const float*)d_in[6];
  const float* enc_Whh1  = (const float*)d_in[7];
  const float* enc_bih1  = (const float*)d_in[8];
  const float* enc_bhh1  = (const float*)d_in[9];
  const float* dec_Wih0  = (const float*)d_in[10];
  const float* dec_Whh0  = (const float*)d_in[11];
  const float* dec_bih0  = (const float*)d_in[12];
  const float* dec_bhh0  = (const float*)d_in[13];
  const float* dec_Wih1  = (const float*)d_in[14];
  const float* dec_Whh1  = (const float*)d_in[15];
  const float* dec_bih1  = (const float*)d_in[16];
  const float* dec_bhh1  = (const float*)d_in[17];
  const float* out1_W    = (const float*)d_in[18];
  const float* out1_b    = (const float*)d_in[19];
  const float* out2_W    = (const float*)d_in[20];

  char* ws = (char*)d_ws;
  size_t off = 0;
  auto bump = [&](size_t bytes) -> void* {
    void* p = ws + off;
    off = (off + bytes + 255) & ~(size_t)255;
    return p;
  };

  unsigned* ctr = (unsigned*)bump(2 * sizeof(unsigned));
  unsigned short* enc_xbf = (unsigned short*)bump((size_t)B_ * TIN_ * INF_ * 2);
  unsigned short* dec_xbf = (unsigned short*)bump((size_t)B_ * TOUT_ * INF_ * 2);
  float* h0f0 = (float*)bump((size_t)B_ * H_ * 4);
  float* h0f1 = (float*)bump((size_t)B_ * H_ * 4);
  float* h1f0 = (float*)bump((size_t)B_ * H_ * 4);
  float* h1f1 = (float*)bump((size_t)B_ * H_ * 4);
  unsigned short* h0b0 = (unsigned short*)bump((size_t)B_ * H_ * 2);
  unsigned short* h0b1 = (unsigned short*)bump((size_t)B_ * H_ * 2);
  unsigned short* h1b0 = (unsigned short*)bump((size_t)B_ * H_ * 2);
  unsigned short* h1b1 = (unsigned short*)bump((size_t)B_ * H_ * 2);
  unsigned short* obf  = (unsigned short*)bump((size_t)B_ * H_ * 2);
  unsigned short* ypb  = (unsigned short*)bump((size_t)B_ * OUTF_ * 2);

  auto pack = [&](const float* W, int N, int K) -> unsigned short* {
    unsigned short* dst =
        (unsigned short*)bump((size_t)(N / 16) * (K / 32) * 512 * 2);
    int threads = (N / 16) * (K / 32) * 32;
    k_pack<<<dim3((threads + 255) / 256), dim3(256), 0, stream>>>(W, dst, N, K);
    return dst;
  };

  k_init<<<dim3(1), dim3(64), 0, stream>>>(ctr);

  {
    int n = B_ * TIN_ * INF_;
    k_cast_bf16<<<dim3((n + 255) / 256), dim3(256), 0, stream>>>(enc_input,
                                                                 enc_xbf, n);
    n = B_ * TOUT_ * INF_;
    k_cast_bf16<<<dim3((n + 255) / 256), dim3(256), 0, stream>>>(dec_input,
                                                                 dec_xbf, n);
  }

  unsigned short* pk_eWih0 = pack(enc_Wih0, 3 * H_, INF_);
  unsigned short* pk_eWhh0 = pack(enc_Whh0, 3 * H_, H_);
  unsigned short* pk_eWih1 = pack(enc_Wih1, 3 * H_, H_);
  unsigned short* pk_eWhh1 = pack(enc_Whh1, 3 * H_, H_);
  unsigned short* pk_dWih0 = pack(dec_Wih0, 3 * H_, INF_ + OUTF_);
  unsigned short* pk_dWhh0 = pack(dec_Whh0, 3 * H_, H_);
  unsigned short* pk_dWih1 = pack(dec_Wih1, 3 * H_, H_);
  unsigned short* pk_dWhh1 = pack(dec_Whh1, 3 * H_, H_);
  unsigned short* pk_out1  = pack(out1_W, H_, H_);
  unsigned short* pk_out2  = pack(out2_W, OUTF_, H_);

  EncParams ep;
  ep.xbf = enc_xbf;
  ep.wih0 = pk_eWih0; ep.whh0 = pk_eWhh0;
  ep.wih1 = pk_eWih1; ep.whh1 = pk_eWhh1;
  ep.bih0 = enc_bih0; ep.bhh0 = enc_bhh0;
  ep.bih1 = enc_bih1; ep.bhh1 = enc_bhh1;
  ep.h0f[0] = h0f0; ep.h0f[1] = h0f1;
  ep.h1f[0] = h1f0; ep.h1f[1] = h1f1;
  ep.h0b[0] = h0b0; ep.h0b[1] = h0b1;
  ep.h1b[0] = h1b0; ep.h1b[1] = h1b1;
  ep.ctr = ctr;
  k_encoder<<<dim3(NWG), dim3(NTHR), 0, stream>>>(ep);

  DecParams dp;
  dp.enc_in = enc_input;
  dp.xbf = dec_xbf;
  dp.wih0 = pk_dWih0; dp.whh0 = pk_dWhh0;
  dp.wih1 = pk_dWih1; dp.whh1 = pk_dWhh1;
  dp.o1pk = pk_out1;  dp.o2pk = pk_out2;
  dp.bih0 = dec_bih0; dp.bhh0 = dec_bhh0;
  dp.bih1 = dec_bih1; dp.bhh1 = dec_bhh1;
  dp.o1b = out1_b;
  dp.h0f[0] = h0f0; dp.h0f[1] = h0f1;
  dp.h1f[0] = h1f0; dp.h1f[1] = h1f1;
  dp.h0b[0] = h0b0; dp.h0b[1] = h0b1;
  dp.h1b[0] = h1b0; dp.h1b[1] = h1b1;
  dp.obf = obf; dp.ypb = ypb;
  dp.out = (float*)d_out;
  dp.ctr = ctr + 1;
  k_decoder<<<dim3(NWG), dim3(NTHR), 0, stream>>>(dp);
}
